// EUCB_18485539242297
// MI455X (gfx1250) — compile-verified, matrix operand order checked by reading, not run on hardware
//
#include <hip/hip_runtime.h>
#include <hip/hip_bf16.h>

typedef __attribute__((ext_vector_type(16))) _Float16 v16h;
typedef __attribute__((ext_vector_type(8)))  _Float16 v8h;
typedef __attribute__((ext_vector_type(8)))  float    v8f;

// ---- Tensor Data Mover availability (probe-verified builtin; arity differs
//      between ROCm 7.2 (clang-22, 5 args) and therock (clang-23, 6 args)) ----
#if defined(__has_builtin)
#  if __has_builtin(__builtin_amdgcn_tensor_load_to_lds) && \
      __has_builtin(__builtin_amdgcn_s_wait_tensorcnt)
#    define USE_TDM 1
#  endif
#endif
#ifndef USE_TDM
#  define USE_TDM 0
#endif

#if USE_TDM
typedef __attribute__((ext_vector_type(4))) unsigned int u32x4;
typedef __attribute__((ext_vector_type(8))) int          i32x8;
typedef __attribute__((ext_vector_type(4))) int          i32x4;

// Issue a 2D TDM tile load (f32 elements): tile [tileY rows][tileX cols] from
// a row-major tensor with row stride stride0 (elements), linearly packed into
// LDS at ldsDst. D# packing per CDNA5 ISA §8 (group0 + group1; groups 2/3 zero
// for <=2D tensors).
__device__ __forceinline__ void tdm_load_2d_f32(const float* src, void* ldsDst,
                                                int tileX, int tileY,
                                                unsigned long long dim0,
                                                unsigned long long dim1,
                                                unsigned long long stride0) {
    const unsigned long long gaddr = (unsigned long long)(const void*)src;
    const unsigned lds = (unsigned)(unsigned long long)ldsDst;  // low 32b = LDS offset
    u32x4 g0;
    g0[0] = 1u;                                        // count=1, is_restore=0, no gather
    g0[1] = lds;                                       // lds_addr
    g0[2] = (unsigned)(gaddr & 0xFFFFFFFFull);         // global_addr[31:0]
    g0[3] = (unsigned)((gaddr >> 32) & 0x01FFFFFFull)  // global_addr[56:32]
          | (2u << 30);                                // type = 2 ("image")
    i32x8 g1;
    g1[0] = (int)(2u << 16);                                           // data_size=4B
    g1[1] = (int)((dim0 & 0xFFFFull) << 16);                           // tensor_dim0[15:0]
    g1[2] = (int)(((dim0 >> 16) & 0xFFFFull) | ((dim1 & 0xFFFFull) << 16));
    g1[3] = (int)(((dim1 >> 16) & 0xFFFFull) | ((unsigned)(tileX & 0xFFFF) << 16));
    g1[4] = (int)(tileY & 0xFFFF);                                     // tile_dim1 (tile_dim2=0)
    g1[5] = (int)(stride0 & 0xFFFFFFFFull);                            // dim0_stride[31:0]
    g1[6] = (int)((stride0 >> 32) & 0xFFFFull);                        // dim0_stride[47:32]
    g1[7] = 0;
    const i32x4 z4 = {0, 0, 0, 0};
#if __clang_major__ >= 23
    const i32x8 z8 = {0, 0, 0, 0, 0, 0, 0, 0};
    __builtin_amdgcn_tensor_load_to_lds(g0, g1, z4, z4, z8, 0);
#else
    __builtin_amdgcn_tensor_load_to_lds(g0, g1, z4, z4, 0);
#endif
}
#endif  // USE_TDM

__device__ __forceinline__ v8f wmma_f16(v16h a, v16h b, v8f c) {
    // 8 args: (neg_a, A, neg_b, B, c_mod, C, reuse_a, reuse_b)
    return __builtin_amdgcn_wmma_f32_16x16x32_f16(false, a, false, b, (short)0, c, false, false);
}

// Fragment lane mapping (16-bit A 16x32 / B 32x16, wave32):
//   lane L: row/col = L&15, K set = {kb..kb+7} u {kb+16..kb+23}, kb = (L>>4)*8
// C/D 16x16 f32: VGPR v -> M = v + 8*(L>>4), N = L&15.

// -------------------------------------------------------------------------
// 1x1 conv as GEMM, LDS-staged. Block = 256 threads (8 waves) computes a
// 16(cout) x 128(pixel) tile; K-loop over 32 input channels per step.
// Activations arrive via TDM (raw f32 tile -> LDS, async, TENSORcnt) when the
// builtin exists, else via float4 vector loads; either way they are converted
// to the transposed f16 [px][ci] fragment layout (rows padded to 40 halves so
// the b128 fragment reads are bank-conflict-free).
// grid = (HW/128, Cout/16, N).
// -------------------------------------------------------------------------
__global__ __launch_bounds__(256)
void conv1x1_wmma(const float* __restrict__ in, const float* __restrict__ w,
                  const float* __restrict__ bias, float* __restrict__ out,
                  int Cin, int Cout, int HW) {
    __shared__ __attribute__((aligned(16))) _Float16 lds_a[128][40];
    __shared__ __attribute__((aligned(16))) _Float16 lds_w[16][40];
#if USE_TDM
    __shared__ __attribute__((aligned(16))) float lds_raw[32][128];
#endif

    const int tid  = threadIdx.x;
    const int lane = tid & 31;
    const int wv   = tid >> 5;                 // wave id 0..7
    const int px0  = blockIdx.x * 128;
    const int co0  = blockIdx.y * 16;
    const int n    = blockIdx.z;
    const float* inN = in + (size_t)n * Cin * HW;

    v8f acc = {};
    for (int k0 = 0; k0 < Cin; k0 += 32) {
        __syncthreads();
        // stage weights: 16co x 32ci (2 elements per thread), coalesced on ci
        {
            int idx = tid;
#pragma unroll
            for (int r = 0; r < 2; ++r, idx += 256) {
                const int ci = idx & 31;
                const int co = idx >> 5;
                lds_w[co][ci] = (_Float16)w[(size_t)(co0 + co) * Cin + (k0 + ci)];
            }
        }
#if USE_TDM
        // async DMA of the raw f32 [32ci][128px] tile into LDS via the TDM
        if (wv == 0) {
            tdm_load_2d_f32(inN + (size_t)k0 * HW + px0, &lds_raw[0][0],
                            /*tileX=*/128, /*tileY=*/32,
                            /*dim0=*/(unsigned long long)HW,
                            /*dim1=*/(unsigned long long)Cin,
                            /*stride0=*/(unsigned long long)HW);
            __builtin_amdgcn_s_wait_tensorcnt(0);
        }
        __syncthreads();
        // LDS->LDS convert + transpose into the f16 fragment layout
        {
            int idx = tid;
#pragma unroll
            for (int r = 0; r < 4; ++r, idx += 256) {
                const int p4 = idx & 31;
                const int ci = idx >> 5;
                const float4 v = *(const float4*)&lds_raw[ci][p4 * 4];
                lds_a[p4 * 4 + 0][ci] = (_Float16)v.x;
                lds_a[p4 * 4 + 1][ci] = (_Float16)v.y;
                lds_a[p4 * 4 + 2][ci] = (_Float16)v.z;
                lds_a[p4 * 4 + 3][ci] = (_Float16)v.w;
            }
        }
#else
        // fallback: stage activations 32ci x 128px via float4 vector loads
        {
            int idx = tid;
#pragma unroll
            for (int r = 0; r < 4; ++r, idx += 256) {
                const int p4 = idx & 31;
                const int ci = idx >> 5;
                const float4 v =
                    *(const float4*)&inN[(size_t)(k0 + ci) * HW + px0 + p4 * 4];
                lds_a[p4 * 4 + 0][ci] = (_Float16)v.x;
                lds_a[p4 * 4 + 1][ci] = (_Float16)v.y;
                lds_a[p4 * 4 + 2][ci] = (_Float16)v.z;
                lds_a[p4 * 4 + 3][ci] = (_Float16)v.w;
                if (k0 + 32 < Cin)
                    __builtin_prefetch(&inN[(size_t)(k0 + 32 + ci) * HW + px0 + p4 * 4], 0, 1);
            }
        }
#endif
        __syncthreads();

        const int m  = lane & 15;
        const int kb = (lane >> 4) << 3;
        const int pl = wv * 16 + m;
        union { v16h v; v8h h[2]; } A, B;
        A.h[0] = *(const v8h*)&lds_w[m][kb];
        A.h[1] = *(const v8h*)&lds_w[m][kb + 16];
        B.h[0] = *(const v8h*)&lds_a[pl][kb];
        B.h[1] = *(const v8h*)&lds_a[pl][kb + 16];
        acc = wmma_f16(A.v, B.v, acc);
    }

    const int col   = lane & 15;
    const int rbase = (lane >> 4) << 3;
    const int px    = px0 + wv * 16 + col;
#pragma unroll
    for (int v = 0; v < 8; ++v) {
        const int co = co0 + v + rbase;
        out[((size_t)n * Cout + co) * HW + px] = acc[v] + bias[co];
    }
}

// -------------------------------------------------------------------------
// 3x3 conv (pad=1) as implicit GEMM, LDS-staged. Block = 8 waves; wave wv
// owns output row h0+wv, a 16-px segment and 16 couts. Per 32-channel K-step
// we stage a 10-row x 18-px halo tile (transposed [row*18+px][ci]) and the
// full 9-tap weight tile [tap][co][ci] (Cout guard staged as zeros).
// Synchronous staging: the halo needs zero-fill at negative offsets, which
// TDM OOB semantics cannot express. Channel-split input pair fuses the concat.
// grid = (W/16, H/8, N*coTiles).
// -------------------------------------------------------------------------
__global__ __launch_bounds__(256)
void conv3x3_wmma(const float* __restrict__ in1, const float* __restrict__ in2,
                  int Csplit, const float* __restrict__ w,
                  const float* __restrict__ bias, float* __restrict__ out,
                  int Cin, int Cout, int H, int W, int coTiles) {
    __shared__ __attribute__((aligned(16))) _Float16 lds_a[180][40];   // [rr*18+pp][ci]
    __shared__ __attribute__((aligned(16))) _Float16 lds_w[9][16][40]; // [tap][co][ci]

    const int tid  = threadIdx.x;
    const int lane = tid & 31;
    const int wv   = tid >> 5;
    const int w0   = blockIdx.x * 16;
    const int h0   = blockIdx.y * 8;
    const int n    = blockIdx.z / coTiles;
    const int co0  = (blockIdx.z % coTiles) * 16;
    const int Crem = Cin - Csplit;
    const size_t HWs = (size_t)H * W;

    v8f acc = {};
    for (int k0 = 0; k0 < Cin; k0 += 32) {
        __syncthreads();
        // stage weights: 9 taps x 16co x 32ci (taps contiguous in memory)
        for (int idx = tid; idx < 9 * 16 * 32; idx += 256) {
            const int tap = idx % 9;
            const int ci  = (idx / 9) & 31;
            const int co  = idx / (9 * 32);
            const float v = (co0 + co < Cout)
                ? w[((size_t)(co0 + co) * Cin + (k0 + ci)) * 9 + tap] : 0.0f;
            lds_w[tap][co][ci] = (_Float16)v;
        }
        // stage halo activation tile: rows h0-1..h0+8, px w0-1..w0+16, 32 ci
        for (int idx = tid; idx < 10 * 18 * 32; idx += 256) {
            const int ci  = idx / 180;
            const int rem = idx % 180;
            const int rr  = rem / 18;
            const int pp  = rem % 18;
            const int hy  = h0 + rr - 1;
            const int wx  = w0 + pp - 1;
            float v = 0.0f;
            if (hy >= 0 && hy < H && wx >= 0 && wx < W) {
                const int cig = k0 + ci;
                const float* src = (cig < Csplit)
                    ? in1 + ((size_t)n * Csplit + cig) * HWs
                    : in2 + ((size_t)n * Crem + (cig - Csplit)) * HWs;
                v = src[(size_t)hy * W + wx];
            }
            lds_a[rr * 18 + pp][ci] = (_Float16)v;
        }
        __syncthreads();

        const int m  = lane & 15;
        const int kb = (lane >> 4) << 3;
#pragma unroll
        for (int ky = 0; ky < 3; ++ky) {
#pragma unroll
            for (int kx = 0; kx < 3; ++kx) {
                const int tap = ky * 3 + kx;
                union { v16h v; v8h h[2]; } A, B;
                A.h[0] = *(const v8h*)&lds_w[tap][m][kb];
                A.h[1] = *(const v8h*)&lds_w[tap][m][kb + 16];
                const int row = (wv + ky) * 18 + m + kx;  // input row h0+wv+ky-1
                B.h[0] = *(const v8h*)&lds_a[row][kb];
                B.h[1] = *(const v8h*)&lds_a[row][kb + 16];
                acc = wmma_f16(A.v, B.v, acc);
            }
        }
    }

    const int col   = lane & 15;
    const int rbase = (lane >> 4) << 3;
    const int h     = h0 + wv;
#pragma unroll
    for (int v = 0; v < 8; ++v) {
        const int co = co0 + v + rbase;
        if (co < Cout)
            out[((size_t)n * Cout + co) * HWs + (size_t)h * W + w0 + col]
                = acc[v] + bias[co];
    }
}

// -------------------------------------------------------------------------
// In-place softmax over the 9 CARAFE taps. kt2 layout [N,36,H,W]; for a given
// sub-pixel r in 0..3 the 9 logits live at channels {4k+r}.
// -------------------------------------------------------------------------
__global__ void softmax9_kernel(float* __restrict__ kt, int N, int HW) {
    const int idx = blockIdx.x * blockDim.x + threadIdx.x;
    const int total = N * HW * 4;
    if (idx >= total) return;
    const int r = idx & 3;
    const int p = (idx >> 2) % HW;
    const int n = (idx >> 2) / HW;
    float* base = kt + ((size_t)n * 36 + r) * HW + p;
    const size_t stride = (size_t)4 * HW;
    float v[9], mx = -1e30f;
#pragma unroll
    for (int k = 0; k < 9; ++k) { v[k] = base[k * stride]; mx = fmaxf(mx, v[k]); }
    float s = 0.0f;
#pragma unroll
    for (int k = 0; k < 9; ++k) { v[k] = __expf(v[k] - mx); s += v[k]; }
    const float inv = 1.0f / s;
#pragma unroll
    for (int k = 0; k < 9; ++k) base[k * stride] = v[k] * inv;
}

// -------------------------------------------------------------------------
// CARAFE reassembly + pixel shuffle fused:
// out[n,c,h2,w2] = sum_k wsm[n,4k+r,h,w] * x1pad[n,c,h+dy,w+dx]
// -------------------------------------------------------------------------
__global__ void carafe_kernel(const float* __restrict__ x1, const float* __restrict__ wsm,
                              float* __restrict__ out, int N, int C, int H, int W) {
    const int H2 = H * 2, W2 = W * 2;
    const long long total = (long long)N * C * H2 * W2;
    const long long idx = (long long)blockIdx.x * blockDim.x + threadIdx.x;
    if (idx >= total) return;
    const int w2 = (int)(idx % W2);
    const int h2 = (int)((idx / W2) % H2);
    const int c  = (int)((idx / ((long long)W2 * H2)) % C);
    const int n  = (int)(idx / ((long long)W2 * H2 * C));
    const int h = h2 >> 1, w = w2 >> 1;
    const int r = ((h2 & 1) << 1) | (w2 & 1);
    const int HW = H * W;
    const float* wp = wsm + ((size_t)n * 36 + r) * HW + (size_t)h * W + w;
    const float* xp = x1 + ((size_t)n * C + c) * HW;
    float s = 0.0f;
#pragma unroll
    for (int dy = -1; dy <= 1; ++dy) {
#pragma unroll
        for (int dx = -1; dx <= 1; ++dx) {
            const int k = (dy + 1) * 3 + (dx + 1);
            const int hh = h + dy, ww = w + dx;
            float xv = 0.0f;
            if (hh >= 0 && hh < H && ww >= 0 && ww < W) xv = xp[(size_t)hh * W + ww];
            s += wp[(size_t)(4 * k) * HW] * xv;
        }
    }
    out[idx] = s;
}

// -------------------------------------------------------------------------
// Depthwise 3x3 (pad=1, no bias).
// -------------------------------------------------------------------------
__global__ void dwconv3x3_kernel(const float* __restrict__ in, const float* __restrict__ w,
                                 float* __restrict__ out, int N, int C, int H, int W) {
    const long long total = (long long)N * C * H * W;
    const long long idx = (long long)blockIdx.x * blockDim.x + threadIdx.x;
    if (idx >= total) return;
    const int x = (int)(idx % W);
    const int y = (int)((idx / W) % H);
    const int c = (int)((idx / ((long long)W * H)) % C);
    const float* src = in + (idx - x - (long long)y * W);
    const float* wc = w + c * 9;
    float s = 0.0f;
#pragma unroll
    for (int ky = 0; ky < 3; ++ky) {
        const int hy = y + ky - 1;
        if (hy < 0 || hy >= H) continue;
#pragma unroll
        for (int kx = 0; kx < 3; ++kx) {
            const int wx = x + kx - 1;
            if (wx < 0 || wx >= W) continue;
            s += wc[ky * 3 + kx] * src[(size_t)hy * W + wx];
        }
    }
    out[idx] = s;
}

// -------------------------------------------------------------------------
// BatchNorm (training-mode, biased var) stats: one block per channel,
// deterministic tree reduction. stats[c] = mean, stats[C+c] = invstd.
// -------------------------------------------------------------------------
__global__ __launch_bounds__(256)
void bn_stats_kernel(const float* __restrict__ x, float* __restrict__ stats,
                     int N, int C, int HW) {
    __shared__ float ssum[256];
    __shared__ float ssq[256];
    const int c = blockIdx.x;
    const int tid = threadIdx.x;
    float s = 0.0f, q = 0.0f;
    for (int n = 0; n < N; ++n) {
        const float* p = x + ((size_t)n * C + c) * HW;
        for (int i = tid; i < HW; i += 256) {
            const float v = p[i];
            s += v; q += v * v;
        }
    }
    ssum[tid] = s; ssq[tid] = q;
    __syncthreads();
    for (int off = 128; off > 0; off >>= 1) {
        if (tid < off) { ssum[tid] += ssum[tid + off]; ssq[tid] += ssq[tid + off]; }
        __syncthreads();
    }
    if (tid == 0) {
        const float cnt = (float)N * (float)HW;
        const float mean = ssum[0] / cnt;
        const float var = ssq[0] / cnt - mean * mean;
        stats[c] = mean;
        stats[C + c] = rsqrtf(var + 1e-5f);
    }
}

__global__ void bn_relu_kernel(float* __restrict__ x, const float* __restrict__ stats,
                               const float* __restrict__ g, const float* __restrict__ b,
                               int C, int HW, long long total) {
    const long long idx = (long long)blockIdx.x * blockDim.x + threadIdx.x;
    if (idx >= total) return;
    const int c = (int)((idx / HW) % C);
    const float v = (x[idx] - stats[c]) * stats[C + c] * g[c] + b[c];
    x[idx] = fmaxf(v, 0.0f);
}

// -------------------------------------------------------------------------
extern "C" void kernel_launch(void* const* d_in, const int* in_sizes, int n_in,
                              void* d_out, int out_size, void* d_ws, size_t ws_size,
                              hipStream_t stream) {
    const float* x1     = (const float*)d_in[0];   // [8,128,64,64]
    const float* x2     = (const float*)d_in[1];   // [8,64,128,128]
    const float* W_down = (const float*)d_in[2];
    const float* b_down = (const float*)d_in[3];
    const float* W_enc  = (const float*)d_in[4];
    const float* b_enc  = (const float*)d_in[5];
    const float* W_cout = (const float*)d_in[6];
    const float* b_cout = (const float*)d_in[7];
    const float* W_dw   = (const float*)d_in[8];
    const float* g1     = (const float*)d_in[9];
    const float* be1    = (const float*)d_in[10];
    const float* W_pwc  = (const float*)d_in[11];
    const float* b_pwc  = (const float*)d_in[12];
    const float* W_c1   = (const float*)d_in[13];
    const float* b_c1   = (const float*)d_in[14];
    const float* g2     = (const float*)d_in[15];
    const float* be2    = (const float*)d_in[16];
    const float* W_c2   = (const float*)d_in[17];
    const float* b_c2   = (const float*)d_in[18];
    const float* g3     = (const float*)d_in[19];
    const float* be3    = (const float*)d_in[20];
    float* out = (float*)d_out;

    const int N = 8, C = 128, Co = 64, H = 64, W = 64, H2 = 128, W2 = 128;
    const int HW1 = H * W;        // 4096
    const int HW2 = H2 * W2;      // 16384

    float* ws = (float*)d_ws;
    float* kt1   = ws;                                  //  8*32*4096   floats
    float* kt2   = kt1 + (size_t)N * 32 * HW1;          //  8*36*4096   floats
    float* stats = kt2 + (size_t)N * 36 * HW1;          //  256 floats
    float* bufC  = stats + 256;                         //  8*128*16384 floats
    float* bufD  = bufC + (size_t)N * C * HW2;          //  8*128*16384 floats

    const long long elemsFull = (long long)N * C * HW2;   // 16,777,216
    const long long elemsHalf = (long long)N * Co * HW2;  //  8,388,608

    // --- CARAFE kernel prediction ---
    // 1) 1x1 down conv: x1 -> kt1 [8,32,64,64]
    conv1x1_wmma<<<dim3(HW1 / 128, 32 / 16, N), 256, 0, stream>>>(
        x1, W_down, b_down, kt1, C, 32, HW1);
    // 2) 3x3 encoder: kt1 -> kt2 [8,36,64,64] (Cout=36 tiled to 48, guarded)
    conv3x3_wmma<<<dim3(W / 16, H / 8, N * 3), 256, 0, stream>>>(
        kt1, kt1, 32, W_enc, b_enc, kt2, 32, 36, H, W, 3);
    // 3) pixel-shuffle + softmax over 9 taps (in place, per sub-pixel r)
    softmax9_kernel<<<(N * HW1 * 4 + 255) / 256, 256, 0, stream>>>(kt2, N, HW1);
    // 4) content-aware reassembly + pixel shuffle -> bufC [8,128,128,128]
    carafe_kernel<<<(unsigned)((elemsFull + 255) / 256), 256, 0, stream>>>(
        x1, kt2, bufC, N, C, H, W);
    // 5) 1x1 out conv: bufC -> bufD (x1_up)
    conv1x1_wmma<<<dim3(HW2 / 128, C / 16, N), 256, 0, stream>>>(
        bufC, W_cout, b_cout, bufD, C, C, HW2);

    // --- EUCB block ---
    // 6) depthwise 3x3: bufD -> bufC
    dwconv3x3_kernel<<<(unsigned)((elemsFull + 255) / 256), 256, 0, stream>>>(
        bufD, W_dw, bufC, N, C, H2, W2);
    // 7) BN(train) + ReLU in place
    bn_stats_kernel<<<C, 256, 0, stream>>>(bufC, stats, N, C, HW2);
    bn_relu_kernel<<<(unsigned)((elemsFull + 255) / 256), 256, 0, stream>>>(
        bufC, stats, g1, be1, C, HW2, elemsFull);
    // 8) channel shuffle (identity) + 1x1 pointwise: bufC -> bufD (x1f, 64ch)
    conv1x1_wmma<<<dim3(HW2 / 128, Co / 16, N), 256, 0, stream>>>(
        bufC, W_pwc, b_pwc, bufD, C, Co, HW2);

    // --- fusion head: concat [x2 ; x1f] handled by split-input conv ---
    // 9) 3x3 conv c1 (Cin=128 split 64/64) -> bufC (64ch)
    conv3x3_wmma<<<dim3(W2 / 16, H2 / 8, N * (Co / 16)), 256, 0, stream>>>(
        x2, bufD, Co, W_c1, b_c1, bufC, C, Co, H2, W2, Co / 16);
    bn_stats_kernel<<<Co, 256, 0, stream>>>(bufC, stats, N, Co, HW2);
    bn_relu_kernel<<<(unsigned)((elemsHalf + 255) / 256), 256, 0, stream>>>(
        bufC, stats, g2, be2, Co, HW2, elemsHalf);
    // 10) 3x3 conv c2 -> d_out
    conv3x3_wmma<<<dim3(W2 / 16, H2 / 8, N * (Co / 16)), 256, 0, stream>>>(
        bufC, bufC, Co, W_c2, b_c2, out, Co, Co, H2, W2, Co / 16);
    bn_stats_kernel<<<Co, 256, 0, stream>>>(out, stats, N, Co, HW2);
    bn_relu_kernel<<<(unsigned)((elemsHalf + 255) / 256), 256, 0, stream>>>(
        out, stats, g3, be3, Co, HW2, elemsHalf);
}
